// XCA_36043365548380
// MI455X (gfx1250) — matrix operand order repaired and checked by reading, not compiled, not run on hardware
//
#include <hip/hip_runtime.h>

// ---------------------------------------------------------------------------
// XCA (cross-covariance attention) for MI455X / gfx1250, bf16 WMMA pipeline.
// B=16, N=4096, C=768, H=16, DH=48
// ---------------------------------------------------------------------------

typedef unsigned short u16;
typedef u16   v8us  __attribute__((ext_vector_type(8)));
typedef u16   v16us __attribute__((ext_vector_type(16)));
typedef __bf16 v16bf __attribute__((ext_vector_type(16)));
typedef float v8f   __attribute__((ext_vector_type(8)));

__device__ __forceinline__ float bf2f(u16 u) {
  unsigned int x = ((unsigned int)u) << 16;
  return __builtin_bit_cast(float, x);
}
__device__ __forceinline__ u16 f2bf(float f) {
  unsigned int u = __builtin_bit_cast(unsigned int, f);
  u += 0x7fffu + ((u >> 16) & 1u);   // round-to-nearest-even
  return (u16)(u >> 16);
}
__device__ __forceinline__ v16us cat16(v8us a, v8us b) {
  return __builtin_shufflevector(a, b, 0,1,2,3,4,5,6,7,8,9,10,11,12,13,14,15);
}
__device__ __forceinline__ v8f wmma_bf16(v16us a, v16us b, v8f c) {
  return __builtin_amdgcn_wmma_f32_16x16x32_bf16(
      false, __builtin_bit_cast(v16bf, a),
      false, __builtin_bit_cast(v16bf, b),
      (short)0, c, false, false);
}

// ---------------------------------------------------------------------------
// 1) f32 -> bf16 convert (grid-stride)
// ---------------------------------------------------------------------------
__global__ void cvt_f32_bf16(const float* __restrict__ src,
                             u16* __restrict__ dst, int n) {
  int i = blockIdx.x * blockDim.x + threadIdx.x;
  int stride = gridDim.x * blockDim.x;
  for (; i < n; i += stride) dst[i] = f2bf(src[i]);
}

// ---------------------------------------------------------------------------
// 2/5) Tiled bf16 WMMA GEMM.  A[M,768] @ B[768,NC].
//   TSTORE=true : epilogue writes bf16 transposed into qkvt [3,16,16,48,4096]
//   TSTORE=false: epilogue writes f32 [M,NC] + bias
// Block: 256 thr (8 waves, 2x4), tile 128x128, K step 32, double-buffered LDS.
// ---------------------------------------------------------------------------
template <int NC, bool TSTORE>
__global__ __launch_bounds__(256) void gemm_bf16_k(
    const u16* __restrict__ A, const u16* __restrict__ Bm,
    u16* __restrict__ outT, float* __restrict__ outF,
    const float* __restrict__ bias) {
  constexpr int K = 768;
  const int m0 = blockIdx.y * 128;
  const int n0 = blockIdx.x * 128;

  __shared__ u16 As[2][128][40];   // [row][k-local], padded
  __shared__ u16 Bt[2][128][40];   // transposed: [col][k-local], padded

  const int tid  = threadIdx.x;
  const int wv   = tid >> 5;
  const int lane = tid & 31;
  const int wrow = wv >> 2;        // 0..1  -> 64 rows each
  const int wcol = wv & 3;         // 0..3  -> 32 cols each
  const int lr   = lane & 15;
  const int lh   = lane >> 4;      // 0/1

  v8f acc[4][2];
  const v8f vz = {0.f,0.f,0.f,0.f,0.f,0.f,0.f,0.f};
#pragma unroll
  for (int mi = 0; mi < 4; mi++)
#pragma unroll
    for (int ni = 0; ni < 2; ni++) acc[mi][ni] = vz;

  auto stage = [&](int buf, int k0) {
    // A tile: 128 x 32 bf16, 512 chunks of 8
    for (int it = tid; it < 512; it += 256) {
      int r = it >> 2, c = (it & 3) << 3;
      const u16* src = A + (size_t)(m0 + r) * K + k0 + c;
      if (k0 + 96 < K) __builtin_prefetch(src + 64, 0, 1);
      *(v8us*)&As[buf][r][c] = *(const v8us*)src;
    }
    // B tile: 32 x 128 bf16, stored transposed into Bt[col][k]
    for (int it = tid; it < 512; it += 256) {
      int kk = it >> 4, c = (it & 15) << 3;
      const u16* src = Bm + (size_t)(k0 + kk) * NC + n0 + c;
      v8us d = *(const v8us*)src;
#pragma unroll
      for (int j = 0; j < 8; j++) Bt[buf][c + j][kk] = d[j];
    }
  };

  stage(0, 0);
  __syncthreads();
  int buf = 0;
  for (int kt = 0; kt < K / 32; kt++) {
    if (kt + 1 < K / 32) stage(buf ^ 1, (kt + 1) * 32);

    v16us afrag[4], bfrag[2];
#pragma unroll
    for (int mi = 0; mi < 4; mi++) {
      const u16* ar = &As[buf][wrow * 64 + mi * 16 + lr][0];
      v8us a0 = *(const v8us*)(ar + lh * 8);
      v8us a1 = *(const v8us*)(ar + 16 + lh * 8);
      afrag[mi] = cat16(a0, a1);
    }
#pragma unroll
    for (int ni = 0; ni < 2; ni++) {
      const u16* br = &Bt[buf][wcol * 32 + ni * 16 + lr][0];
      v8us b0 = *(const v8us*)(br + lh * 8);
      v8us b1 = *(const v8us*)(br + 16 + lh * 8);
      bfrag[ni] = cat16(b0, b1);
    }
#pragma unroll
    for (int mi = 0; mi < 4; mi++)
#pragma unroll
      for (int ni = 0; ni < 2; ni++)
        acc[mi][ni] = wmma_bf16(afrag[mi], bfrag[ni], acc[mi][ni]);

    __syncthreads();
    buf ^= 1;
  }

  if constexpr (TSTORE) {
    // qkvt[which][b][h][d][n]  (bf16), per-lane 8 consecutive tokens
#pragma unroll
    for (int mi = 0; mi < 4; mi++)
#pragma unroll
      for (int ni = 0; ni < 2; ni++) {
        int col = n0 + wcol * 32 + ni * 16 + lr;      // 0..2303
        int which = col / 768;
        int rem = col - which * 768;
        int hh = rem / 48, dd = rem - hh * 48;
        int mrow = m0 + wrow * 64 + mi * 16 + lh * 8;
        int bidx = mrow >> 12;                        // /4096
        int ntok = mrow & 4095;
        u16* dst = outT +
          ((((size_t)which * 16 + bidx) * 16 + hh) * 48 + dd) * 4096 + ntok;
        v8us pk;
#pragma unroll
        for (int i = 0; i < 8; i++) pk[i] = f2bf(acc[mi][ni][i]);
        *(v8us*)dst = pk;
      }
  } else {
#pragma unroll
    for (int mi = 0; mi < 4; mi++)
#pragma unroll
      for (int ni = 0; ni < 2; ni++) {
        int col = n0 + wcol * 32 + ni * 16 + lr;
        float bv = bias[col];
        int row0 = m0 + wrow * 64 + mi * 16 + lh * 8;
#pragma unroll
        for (int i = 0; i < 8; i++)
          outF[(size_t)(row0 + i) * NC + col] = acc[mi][ni][i] + bv;
      }
  }
}

// ---------------------------------------------------------------------------
// 3) Per-head: L2 norms over tokens, attn = (q/||q||)(k/||k||)^T * temp,
//    softmax over e, emit bf16 attn padded to K=64.
//    Block = 288 thr = 9 waves = the 9 16x16 tiles of the 48x48 Gram matrix.
// ---------------------------------------------------------------------------
__global__ __launch_bounds__(288) void attn_softmax_k(
    const u16* __restrict__ qkvt, const float* __restrict__ temperature,
    u16* __restrict__ attnbf) {
  const int head = blockIdx.x;                 // b*16 + h
  const u16* qh = qkvt + (size_t)(0 * 256 + head) * 48 * 4096;
  const u16* kh = qkvt + (size_t)(1 * 256 + head) * 48 * 4096;

  __shared__ float inv[2][48];
  __shared__ float attnS[48][49];

  const int wv = threadIdx.x >> 5;
  const int lane = threadIdx.x & 31;
  const int lr = lane & 15, lh = lane >> 4;

  // --- row norms: 96 rows (q then k), wave-parallel, shuffle reduction ---
  for (int r = wv; r < 96; r += 9) {
    const u16* row = (r < 48) ? (qh + (size_t)r * 4096)
                              : (kh + (size_t)(r - 48) * 4096);
    float s = 0.f;
    for (int j = lane; j < 512; j += 32) {
      v8us d = *(const v8us*)(row + j * 8);
#pragma unroll
      for (int e = 0; e < 8; e++) { float f = bf2f(d[e]); s += f * f; }
    }
#pragma unroll
    for (int m = 16; m; m >>= 1) s += __shfl_xor(s, m, 32);
    if (lane == 0)
      inv[r / 48][r % 48] = 1.0f / fmaxf(sqrtf(s), 1e-12f);
  }
  __syncthreads();

  // --- q @ k^T tile via WMMA, K = 4096 tokens ---
  const int mi = wv / 3, ni = wv % 3;
  v8f acc = {0.f,0.f,0.f,0.f,0.f,0.f,0.f,0.f};
  const u16* arow = qh + (size_t)(mi * 16 + lr) * 4096;
  const u16* brow = kh + (size_t)(ni * 16 + lr) * 4096;
  for (int kb = 0; kb < 4096; kb += 32) {
    v8us a0 = *(const v8us*)(arow + kb + lh * 8);
    v8us a1 = *(const v8us*)(arow + kb + 16 + lh * 8);
    v8us b0 = *(const v8us*)(brow + kb + lh * 8);
    v8us b1 = *(const v8us*)(brow + kb + 16 + lh * 8);
    acc = wmma_bf16(cat16(a0, a1), cat16(b0, b1), acc);
  }

  const float tmp = temperature[head & 15];
  const int col = ni * 16 + lr;
  const int row0 = mi * 16 + lh * 8;
  const float ivk = inv[1][col];
#pragma unroll
  for (int i = 0; i < 8; i++)
    attnS[row0 + i][col] = acc[i] * inv[0][row0 + i] * ivk * tmp;
  __syncthreads();

  // --- softmax over e (48 wide) per row, write bf16 padded to 64 ---
  for (int r = wv; r < 48; r += 9) {
    float v1 = attnS[r][lane];
    float v2 = (lane < 16) ? attnS[r][lane + 32] : -__builtin_inff();
    float mx = fmaxf(v1, v2);
#pragma unroll
    for (int m = 16; m; m >>= 1) mx = fmaxf(mx, __shfl_xor(mx, m, 32));
    float e1 = __expf(v1 - mx);
    float e2 = (lane < 16) ? __expf(v2 - mx) : 0.f;
    float s = e1 + e2;
#pragma unroll
    for (int m = 16; m; m >>= 1) s += __shfl_xor(s, m, 32);
    float is = 1.0f / s;
    u16* dst = attnbf + (size_t)head * 48 * 64 + r * 64;
    dst[lane] = f2bf(e1 * is);
    dst[lane + 32] = (lane < 16) ? f2bf(e2 * is) : (u16)0;  // zero-pad 48..63
  }
}

// ---------------------------------------------------------------------------
// 4) out = attn @ v : per head, M=48(d) x K=48->64(e) x N=4096 tokens.
//    Block = 256 thr (8 waves x 16 tokens = 128-token tile), v transposed
//    through LDS for contiguous B fragments. Stores bf16 into [B,N,C] layout.
// ---------------------------------------------------------------------------
__global__ __launch_bounds__(256) void attn_v_k(
    const u16* __restrict__ qkvt, const u16* __restrict__ attnbf,
    u16* __restrict__ yb) {
  const int head = blockIdx.y;
  const int t0 = blockIdx.x * 128;
  const u16* vh = qkvt + (size_t)(2 * 256 + head) * 48 * 4096;

  __shared__ u16 Vt[128][72];    // [token][e], e padded 64->72

  // stage v tile transposed (coalesced global reads, scattered LDS writes)
  for (int it = threadIdx.x; it < 768; it += 256) {
    int e = it >> 4, ck = it & 15;
    v8us d = *(const v8us*)(vh + (size_t)e * 4096 + t0 + ck * 8);
#pragma unroll
    for (int j = 0; j < 8; j++) Vt[ck * 8 + j][e] = d[j];
  }
  for (int it = threadIdx.x; it < 128 * 16; it += 256) {
    Vt[it >> 4][48 + (it & 15)] = 0;    // zero rows e = 48..63
  }
  __syncthreads();

  const int wv = threadIdx.x >> 5;
  const int lane = threadIdx.x & 31;
  const int lr = lane & 15, lh = lane >> 4;

  const u16* abase = attnbf + (size_t)head * 48 * 64;
  const v8f vz = {0.f,0.f,0.f,0.f,0.f,0.f,0.f,0.f};
  v8f acc[3] = {vz, vz, vz};

  for (int kb = 0; kb < 64; kb += 32) {
    const u16* br = &Vt[wv * 16 + lr][kb + lh * 8];
    v16us bv = cat16(*(const v8us*)br, *(const v8us*)(br + 16));
#pragma unroll
    for (int mi = 0; mi < 3; mi++) {
      const u16* ar = abase + (size_t)(mi * 16 + lr) * 64;
      v8us a0 = *(const v8us*)(ar + kb + lh * 8);
      v8us a1 = *(const v8us*)(ar + kb + 16 + lh * 8);
      acc[mi] = wmma_bf16(cat16(a0, a1), bv, acc[mi]);
    }
  }

  const int b = head >> 4, h = head & 15;
  const int ntok = t0 + wv * 16 + lr;
#pragma unroll
  for (int mi = 0; mi < 3; mi++) {
    int d0 = mi * 16 + lh * 8;
    u16* dst = yb + ((size_t)b * 4096 + ntok) * 768 + h * 48 + d0;
    v8us pk;
#pragma unroll
    for (int i = 0; i < 8; i++) pk[i] = f2bf(acc[mi][i]);
    *(v8us*)dst = pk;
  }
}

// ---------------------------------------------------------------------------
// launch
// ---------------------------------------------------------------------------
extern "C" void kernel_launch(void* const* d_in, const int* in_sizes, int n_in,
                              void* d_out, int out_size, void* d_ws,
                              size_t ws_size, hipStream_t stream) {
  const float* x           = (const float*)d_in[0];   // [16,4096,768]
  const float* w_qkv       = (const float*)d_in[1];   // [768,2304]
  const float* temperature = (const float*)d_in[2];   // [16,1,1]
  const float* w_proj      = (const float*)d_in[3];   // [768,768]
  const float* b_proj      = (const float*)d_in[4];   // [768]
  float* out = (float*)d_out;                         // [16,4096,768]

  char* ws = (char*)d_ws;
  size_t off = 0;
  auto take = [&](size_t bytes) -> void* {
    off = (off + 255) & ~(size_t)255;
    void* p = ws + off;
    off += bytes;
    return p;
  };

  const size_t M = 65536;  // B*N
  u16* xb     = (u16*)take(M * 768 * 2);
  u16* wqkvb  = (u16*)take((size_t)768 * 2304 * 2);
  u16* wprojb = (u16*)take((size_t)768 * 768 * 2);
  u16* qkvt   = (u16*)take((size_t)3 * 256 * 48 * 4096 * 2);
  u16* attnbf = (u16*)take((size_t)256 * 48 * 64 * 2);
  u16* yb     = (u16*)take(M * 768 * 2);
  (void)ws_size; (void)in_sizes; (void)n_in; (void)out_size;

  cvt_f32_bf16<<<4096, 256, 0, stream>>>(x, xb, (int)(M * 768));
  cvt_f32_bf16<<<512, 256, 0, stream>>>(w_qkv, wqkvb, 768 * 2304);
  cvt_f32_bf16<<<256, 256, 0, stream>>>(w_proj, wprojb, 768 * 768);

  // GEMM1: x @ w_qkv -> qkvt (transposed bf16 [3,B,H,DH,N])
  gemm_bf16_k<2304, true><<<dim3(18, 512), 256, 0, stream>>>(
      xb, wqkvb, qkvt, nullptr, nullptr);

  // per-head normalize + Gram + softmax
  attn_softmax_k<<<256, 288, 0, stream>>>(qkvt, temperature, attnbf);

  // attn @ v -> yb ([B,N,C] bf16)
  attn_v_k<<<dim3(32, 256), 256, 0, stream>>>(qkvt, attnbf, yb);

  // GEMM2: yb @ w_proj + b_proj -> out (f32)
  gemm_bf16_k<768, false><<<dim3(6, 512), 256, 0, stream>>>(
      yb, wprojb, nullptr, out, b_proj);
}
